// SWEMCore_62869731279657
// MI455X (gfx1250) — matrix operand order repaired and checked by reading, not compiled, not run on hardware
//
#include <hip/hip_runtime.h>
#include <math.h>

// ---------------------------------------------------------------------------
// SWEM pipeline for MI455X (gfx1250, wave32, WMMA bf16 16x16x32).
// All heavy GEMMs run on v_wmma_f32_16x16x32_bf16 with vectorized fragment
// loads (b128). The dominant 3x3 gated conv is im2col + GEMM with the B tile
// double-buffered into LDS via GLOBAL_LOAD_ASYNC_TO_LDS_B128 (ASYNCcnt),
// shared by all 8 waves of the block.
// ---------------------------------------------------------------------------

typedef __bf16 bf16_t;
typedef __attribute__((ext_vector_type(16))) __bf16 v16bf;
typedef __attribute__((ext_vector_type(8)))  __bf16 v8bf;
typedef __attribute__((ext_vector_type(8)))  float  v8f;
typedef __attribute__((ext_vector_type(4)))  float  v4f;

// ---- problem constants (match reference) ----------------------------------
#define DB   2
#define DN   4
#define CK   64
#define CV   512
#define DH   64
#define DW   64
#define HW   4096
#define DL   256
#define TOPL 64
#define CIN  1152          // 2*Cv + 2*TOPL
#define KTOT (CIN * 9)     // 10368, conv GEMM K
#define BN   (DB * DN)     // 8
#define BNS  (DB * DN * 2) // 16
#define INV_TAU 20.0f

// ---- workspace layout (bytes, 256-aligned). needs ~210 MB of d_ws. --------
#define AL(x) (((x) + 255) & ~(size_t)255)
static const size_t OFF_XT    = 0;                                   // bf16 (B,HW,CK)  x_t
static const size_t SZ_XT     = AL((size_t)DB * HW * CK * 2);
static const size_t OFF_XTN   = OFF_XT + SZ_XT;                      // bf16 (B,HW,CK)  l2norm
static const size_t OFF_XB    = OFF_XTN + SZ_XT;                     // bf16 (B,CK,HW)  x
static const size_t OFF_KNT   = OFF_XB + SZ_XT;                      // bf16 (BNS,L,CK) kn^T
static const size_t SZ_KNT    = AL((size_t)BNS * DL * CK * 2);
static const size_t OFF_KAPPA = OFF_KNT + SZ_KNT;                    // f32 (BNS,CK,L)
static const size_t SZ_KAPPA  = AL((size_t)BNS * CK * DL * 4);
static const size_t OFF_WEI   = OFF_KAPPA + SZ_KAPPA;                // f32 (BNS,HW)
static const size_t SZ_WEI    = AL((size_t)BNS * HW * 4);
static const size_t OFF_ZITA  = OFF_WEI + SZ_WEI;                    // f32 (BNS,L)
static const size_t SZ_ZITA   = AL((size_t)BNS * DL * 4);
static const size_t OFF_RAW   = OFF_ZITA + SZ_ZITA;                  // f32 (zz_raw / s / aff / exp_aff)
static const size_t SZ_RAW    = AL((size_t)BNS * HW * DL * 4);
static const size_t OFF_ZZB   = OFF_RAW + SZ_RAW;                    // bf16 zz^T (BNS,L,HW) OR paff^T (BN,HW,2L)
static const size_t SZ_ZZB    = AL((size_t)BNS * HW * DL * 2);
static const size_t OFF_MV2   = OFF_ZZB + SZ_ZZB;                    // bf16 (BN,CV,2L)
static const size_t SZ_MV2    = AL((size_t)BN * CV * 2 * DL * 2);
static const size_t OFF_FIN   = OFF_MV2 + SZ_MV2;                    // bf16 (BN,CIN,HW)
static const size_t SZ_FIN    = AL((size_t)BN * CIN * HW * 2);
static const size_t OFF_WFB   = OFF_FIN + SZ_FIN;                    // bf16 (CV,KTOT)
static const size_t SZ_WB     = AL((size_t)CV * KTOT * 2);
static const size_t OFF_WAB   = OFF_WFB + SZ_WB;
// im2col B^T (HW,KTOT) bf16 = ~85MB: overlays RAW+ZZB (dead by conv time)
static const size_t OFF_IM2   = OFF_RAW;

// ---------------------------------------------------------------------------
// WMMA fragment helpers (ISA 7.12.2 layouts, wave32) — vectorized loads
// ---------------------------------------------------------------------------
union frag16 { v16bf v; struct { v8bf lo, hi; } h; };

// A 16x32 bf16 from row-major (M,K): lane (half,row) -> two contiguous 16B chunks
__device__ __forceinline__ v16bf load_a_rm(const bf16_t* A, int lda, int lane) {
  int half = lane >> 4, row = lane & 15;
  const bf16_t* p = A + (size_t)row * lda + half * 8;
  frag16 u;
  u.h.lo = *(const v8bf*)p;        // k = half*8 + 0..7
  u.h.hi = *(const v8bf*)(p + 16); // k = 16 + half*8 + 0..7
  return u.v;
}
// A from row-major f32 with convert (qv_mem)
__device__ __forceinline__ v16bf load_a_rm_f32(const float* A, int lda, int lane) {
  int half = lane >> 4, row = lane & 15;
  const float* p = A + (size_t)row * lda + half * 8;
  v4f a0 = *(const v4f*)p,        a1 = *(const v4f*)(p + 4);
  v4f b0 = *(const v4f*)(p + 16), b1 = *(const v4f*)(p + 20);
  v16bf f;
#pragma unroll
  for (int i = 0; i < 4; ++i) {
    f[i] = (bf16_t)a0[i]; f[4 + i]  = (bf16_t)a1[i];
    f[8 + i] = (bf16_t)b0[i]; f[12 + i] = (bf16_t)b1[i];
  }
  return f;
}
// B 32x16 bf16 from *transposed* storage B^T (N,K) row-major:
// lane (kh,col) -> K = kh*8 + 0..7 and 16 + kh*8 + 0..7 at Bt[col*ldb + ...]
__device__ __forceinline__ v16bf load_b_tn(const bf16_t* Bt, int ldb, int lane) {
  int col = lane & 15, kh = lane >> 4;
  const bf16_t* p = Bt + (size_t)col * ldb + kh * 8;
  frag16 u;
  u.h.lo = *(const v8bf*)p;
  u.h.hi = *(const v8bf*)(p + 16);
  return u.v;
}
// B fragment from LDS chunk (64 rows x 32 k, row-major, K-contiguous)
__device__ __forceinline__ v16bf load_b_lds(const bf16_t* L, int lane) {
  int col = lane & 15, kh = lane >> 4;
  const bf16_t* p = L + col * 32 + kh * 8;
  frag16 u;
  u.h.lo = *(const v8bf*)p;
  u.h.hi = *(const v8bf*)(p + 16);
  return u.v;
}
__device__ __forceinline__ v8f wmma_bf16(v16bf a, v16bf b, v8f c) {
  return __builtin_amdgcn_wmma_f32_16x16x32_bf16(false, a, false, b, (short)0, c, false, false);
}
__device__ __forceinline__ void store_c_f32(float* C, int ldc, int lane, v8f acc) {
  int col = lane & 15, mh = lane >> 4;
#pragma unroll
  for (int v = 0; v < 8; ++v) C[(size_t)(mh * 8 + v) * ldc + col] = acc[v];
}
// Async global->LDS copy of 16 bytes per lane (GLOBAL_LOAD_ASYNC_TO_LDS_B128,
// tracked by ASYNCcnt). Inline asm: portable across both gfx1250 toolchains.
__device__ __forceinline__ void async_copy16(unsigned lds_off, const bf16_t* gptr) {
  asm volatile("global_load_async_to_lds_b128 %0, %1, off"
               :: "v"(lds_off), "v"((unsigned long long)(size_t)gptr)
               : "memory");
}
__device__ __forceinline__ void wait_async0() {
  asm volatile("s_wait_asynccnt 0" ::: "memory");
}

// ---------------------------------------------------------------------------
// K0: prep — bf16 copies of qk (x, x_t) and l2norm over Ck (x_t_n == qk_n)
// ---------------------------------------------------------------------------
__global__ void k_prep(const float* __restrict__ qk, bf16_t* xt, bf16_t* xtn, bf16_t* xb) {
  int idx = blockIdx.x * blockDim.x + threadIdx.x;
  if (idx >= DB * HW) return;
  int b = idx / HW, hw = idx % HW;
  float ss = 0.f;
#pragma unroll 4
  for (int c = 0; c < CK; ++c) {
    float v = qk[((size_t)b * CK + c) * HW + hw];
    ss += v * v;
  }
  float inv = 1.f / (sqrtf(ss) + 1e-6f);
#pragma unroll 4
  for (int c = 0; c < CK; ++c) {
    float v = qk[((size_t)b * CK + c) * HW + hw];
    xt [((size_t)b * HW + hw) * CK + c] = (bf16_t)v;
    xtn[((size_t)b * HW + hw) * CK + c] = (bf16_t)(v * inv);
    xb [((size_t)b * CK + c) * HW + hw] = (bf16_t)v;
  }
}

__global__ void k_copy_f32(const float* __restrict__ src, float* dst, int n) {
  int i = blockIdx.x * blockDim.x + threadIdx.x;
  if (i < n) dst[i] = src[i];
}

// ---------------------------------------------------------------------------
// K1: knT(bns,l,ck) = l2norm(kappa, axis=Ck) -> bf16, K-contiguous.
// ---------------------------------------------------------------------------
__global__ void k_knorm(const float* __restrict__ kappa, bf16_t* knt) {
  int idx = blockIdx.x * blockDim.x + threadIdx.x;
  if (idx >= BNS * DL) return;
  int bns = idx / DL, l = idx % DL;
  size_t base = (size_t)bns * CK * DL + l;
  float ss = 0.f;
#pragma unroll 4
  for (int c = 0; c < CK; ++c) { float v = kappa[base + (size_t)c * DL]; ss += v * v; }
  float inv = 1.f / (sqrtf(ss) + 1e-6f);
  bf16_t* o = knt + ((size_t)bns * DL + l) * CK;
#pragma unroll 4
  for (int c = 0; c < CK; ++c)
    o[c] = (bf16_t)(kappa[base + (size_t)c * DL] * inv);
}

// ---------------------------------------------------------------------------
// K2: stage-1 GEMM  out(bns,HW,L) = x_t(HW,CK) @ kn(CK,L)    K=64
// ---------------------------------------------------------------------------
__global__ __launch_bounds__(256) void k_gemm1(const bf16_t* __restrict__ Abase,
                                               const bf16_t* __restrict__ knt,
                                               float* __restrict__ out) {
  int wave = threadIdx.x >> 5, lane = threadIdx.x & 31;
  int bns = blockIdx.y;
  int g = blockIdx.x * 8 + wave;          // 0..1023
  int mt = g >> 2, ns = g & 3;            // 256 M-tiles, 4 N-strips of 64
  int b = bns >> 3;
  const bf16_t* A  = Abase + (size_t)b * HW * CK + (size_t)mt * 16 * CK;
  const bf16_t* Bt = knt + (size_t)bns * DL * CK + (size_t)ns * 64 * CK;
  v8f acc[4] = {};
#pragma unroll
  for (int kk = 0; kk < CK; kk += 32) {
    v16bf af = load_a_rm(A + kk, CK, lane);
#pragma unroll
    for (int t = 0; t < 4; ++t)
      acc[t] = wmma_bf16(af, load_b_tn(Bt + (size_t)t * 16 * CK + kk, CK, lane), acc[t]);
  }
  float* O = out + (size_t)bns * HW * DL + (size_t)mt * 16 * DL + ns * 64;
#pragma unroll
  for (int t = 0; t < 4; ++t) store_c_f32(O + t * 16, DL, lane, acc[t]);
}

// ---------------------------------------------------------------------------
// K3: zz^T(bns,l,hw) = softmax(raw/tau over L) * weights  -> bf16.
// ---------------------------------------------------------------------------
__global__ __launch_bounds__(256) void k_softmax_zz(const float* __restrict__ raw,
                                                    const float* __restrict__ wei,
                                                    bf16_t* __restrict__ zzt) {
  int g = blockIdx.x * 256 + threadIdx.x;   // 0..BNS*HW-1
  int bns = g >> 12, hw = g & (HW - 1);
  const float* r = raw + ((size_t)bns * HW + hw) * DL;
  float mx = -3.4e38f;
  for (int l = 0; l < DL; ++l) mx = fmaxf(mx, r[l]);
  float sum = 0.f;
  for (int l = 0; l < DL; ++l) sum += __expf((r[l] - mx) * INV_TAU);
  float w = wei[g] / sum;
  bf16_t* o = zzt + (size_t)bns * DL * HW + hw;
  for (int l = 0; l < DL; ++l)
    o[(size_t)l * HW] = (bf16_t)(w * __expf((r[l] - mx) * INV_TAU));
}

// ---------------------------------------------------------------------------
// K4: zita(bns,l) = zita0 + sum_hw zz^T[l][hw] (contiguous). grid(L,BNS).
// ---------------------------------------------------------------------------
__global__ void k_zita(const bf16_t* __restrict__ zzt, const float* __restrict__ zita0,
                       float* __restrict__ zita) {
  __shared__ float sm[256];
  int l = blockIdx.x, bns = blockIdx.y, t = threadIdx.x;
  const bf16_t* r = zzt + ((size_t)bns * DL + l) * HW;
  float s = 0.f;
  for (int hw = t; hw < HW; hw += 256) s += (float)r[hw];
  sm[t] = s; __syncthreads();
  for (int o = 128; o; o >>= 1) { if (t < o) sm[t] += sm[t + o]; __syncthreads(); }
  if (t == 0) zita[bns * DL + l] = zita0[bns * DL + l] + sm[0];
}

// ---------------------------------------------------------------------------
// K5: kappa = (zita0*kappa0 + x @ zz) / zita.  M=64 N=256 K=4096 per bns.
// ---------------------------------------------------------------------------
__global__ __launch_bounds__(256) void k_kappa(const bf16_t* __restrict__ xb,
                                               const bf16_t* __restrict__ zzt,
                                               const float* __restrict__ kappa0,
                                               const float* __restrict__ zita0,
                                               const float* __restrict__ zita,
                                               float* __restrict__ kappa) {
  int wave = threadIdx.x >> 5, lane = threadIdx.x & 31;
  int bns = blockIdx.y;
  int tile = blockIdx.x * 8 + wave;       // 0..63
  int mt = tile >> 4, nt = tile & 15;
  int b = bns >> 3;
  const bf16_t* A  = xb + (size_t)b * CK * HW + (size_t)mt * 16 * HW;
  const bf16_t* Bt = zzt + (size_t)bns * DL * HW + (size_t)nt * 16 * HW;
  v8f acc = {};
  for (int kk = 0; kk < HW; kk += 32)
    acc = wmma_bf16(load_a_rm(A + kk, HW, lane), load_b_tn(Bt + kk, HW, lane), acc);
  int col = lane & 15, mh = lane >> 4;
#pragma unroll
  for (int v = 0; v < 8; ++v) {
    int m = mt * 16 + mh * 8 + v, l = nt * 16 + col;
    size_t idx = (size_t)bns * CK * DL + (size_t)m * DL + l;
    kappa[idx] = (zita0[bns * DL + l] * kappa0[idx] + acc[v]) / zita[bns * DL + l];
  }
}

// ---------------------------------------------------------------------------
// K6: sww weight update. one thread per (b,n,hw); max over (s,L), props.
// ---------------------------------------------------------------------------
__global__ __launch_bounds__(256) void k_sww(const float* __restrict__ s_raw,
                                             const float* __restrict__ masks,
                                             float* __restrict__ wei) {
  int g = blockIdx.x * 256 + threadIdx.x;   // 0..BN*HW-1
  int hw = g & (HW - 1), bn = g >> 12;
  const float* r0 = s_raw + ((size_t)(bn * 2 + 0) * HW + hw) * DL;
  const float* r1 = s_raw + ((size_t)(bn * 2 + 1) * HW + hw) * DL;
  float mx = -3.4e38f;
  for (int l = 0; l < DL; ++l) { mx = fmaxf(mx, r0[l]); mx = fmaxf(mx, r1[l]); }
  float s0 = 0.f, s1 = 0.f;
  for (int l = 0; l < DL; ++l) {
    s0 += __expf((r0[l] - mx) * INV_TAU);
    s1 += __expf((r1[l] - mx) * INV_TAU);
  }
  float inv = 1.f / (s0 + s1);
  size_t w0 = (size_t)(bn * 2 + 0) * HW + hw, w1 = (size_t)(bn * 2 + 1) * HW + hw;
  wei[w0] = masks[w0] * (1.f - s0 * inv);
  wei[w1] = masks[w1] * (1.f - s1 * inv);
}

// ---------------------------------------------------------------------------
// K7: nu GEMM -> mv2 bf16. M=512 N=256 K=4096 per bns. grid(64, BNS).
// ---------------------------------------------------------------------------
__global__ __launch_bounds__(256) void k_nu(const float* __restrict__ qv_mem,
                                            const bf16_t* __restrict__ zzt,
                                            const float* __restrict__ zita,
                                            bf16_t* __restrict__ mv2) {
  int wave = threadIdx.x >> 5, lane = threadIdx.x & 31;
  int bns = blockIdx.y;
  int tile = blockIdx.x * 8 + wave;       // 0..511
  int mt = tile >> 4, nt = tile & 15;
  int bn = bns >> 1, s = bns & 1;
  const float*  A  = qv_mem + ((size_t)bn * CV + (size_t)mt * 16) * HW;
  const bf16_t* Bt = zzt + (size_t)bns * DL * HW + (size_t)nt * 16 * HW;
  v8f acc = {};
  for (int kk = 0; kk < HW; kk += 32)
    acc = wmma_bf16(load_a_rm_f32(A + kk, HW, lane), load_b_tn(Bt + kk, HW, lane), acc);
  int col = lane & 15, mh = lane >> 4;
#pragma unroll
  for (int v = 0; v < 8; ++v) {
    int m = mt * 16 + mh * 8 + v, l = nt * 16 + col;
    mv2[((size_t)bn * CV + m) * (2 * DL) + s * DL + l] =
        (bf16_t)(acc[v] / zita[bns * DL + l]);
  }
}

// ---------------------------------------------------------------------------
// K8: affinity GEMM  aff(bns,L,HW) = kn^T(L,CK) @ qk_n(CK,HW).  K=64.
// ---------------------------------------------------------------------------
__global__ __launch_bounds__(256) void k_aff(const bf16_t* __restrict__ knt,
                                             const bf16_t* __restrict__ xtn,
                                             float* __restrict__ out) {
  int wave = threadIdx.x >> 5, lane = threadIdx.x & 31;
  int bns = blockIdx.y;
  int g = blockIdx.x * 8 + wave;          // 0..1023
  int mt = g >> 6, ns = g & 63;           // 16 M-tiles, 64 N-strips of 64
  int b = bns >> 3;
  const bf16_t* A = knt + (size_t)bns * DL * CK + (size_t)mt * 16 * CK;
  v8f acc[4] = {};
#pragma unroll
  for (int kk = 0; kk < CK; kk += 32) {
    v16bf af = load_a_rm(A + kk, CK, lane);
#pragma unroll
    for (int t = 0; t < 4; ++t) {
      const bf16_t* Bt = xtn + ((size_t)b * HW + ns * 64 + t * 16) * CK + kk;
      acc[t] = wmma_bf16(af, load_b_tn(Bt, CK, lane), acc[t]);
    }
  }
  float* O = out + (size_t)bns * DL * HW + (size_t)mt * 16 * HW + ns * 64;
#pragma unroll
  for (int t = 0; t < 4; ++t) store_c_f32(O + t * 16, HW, lane, acc[t]);
}

// ---------------------------------------------------------------------------
// K9: p_aff softmax over (s,L) per (bn,hw); exp in place + paff^T bf16.
// ---------------------------------------------------------------------------
__global__ __launch_bounds__(256) void k_paff(float* __restrict__ raw,
                                              bf16_t* __restrict__ pafft) {
  int bn = blockIdx.y;
  int hw = blockIdx.x * 256 + threadIdx.x;
  size_t base = (size_t)bn * 2 * DL * HW + hw;
  float mx = -3.4e38f;
  for (int j = 0; j < 2 * DL; ++j) mx = fmaxf(mx, raw[base + (size_t)j * HW]);
  float sum = 0.f;
  for (int j = 0; j < 2 * DL; ++j) {
    float e = __expf((raw[base + (size_t)j * HW] - mx) * INV_TAU);
    raw[base + (size_t)j * HW] = e;       // exp_aff kept for top-k
    sum += e;
  }
  float inv = 1.f / sum;
  bf16_t* o = pafft + ((size_t)bn * HW + hw) * (2 * DL);
  for (int j = 0; j < 2 * DL; ++j)
    o[j] = (bf16_t)(raw[base + (size_t)j * HW] * inv);
}

// ---------------------------------------------------------------------------
// K10: top-64-of-256 per (bn,hw,s) + cumsum ratio -> S channels of fin.
// ---------------------------------------------------------------------------
__global__ __launch_bounds__(16) void k_topk(const float* __restrict__ ea,
                                             bf16_t* __restrict__ fin) {
  __shared__ float sv[16 * 512];
  __shared__ float c0[16 * 64];
  int t = threadIdx.x;
  int task = blockIdx.x * 16 + t;         // 0..BN*HW-1
  int hw = task & 4095, bn = task >> 12;
  float* v = sv + t * 512;
  for (int j = 0; j < 512; ++j)
    v[j] = ea[((size_t)(bn * 2 + (j >> 8)) * DL + (j & 255)) * HW + hw];
  float run = 0.f;
  for (int i = 0; i < TOPL; ++i) {        // s = 0
    float best = -3.4e38f; int bi = 0;
    for (int j = 0; j < 256; ++j) if (v[j] > best) { best = v[j]; bi = j; }
    run += best; v[bi] = -3.4e38f;
    c0[t * 64 + i] = run;
  }
  run = 0.f;
  for (int i = 0; i < TOPL; ++i) {        // s = 1, emit Sf on the fly
    float best = -3.4e38f; int bi = 256;
    for (int j = 256; j < 512; ++j) if (v[j] > best) { best = v[j]; bi = j; }
    run += best; v[bi] = -3.4e38f;
    float f0 = c0[t * 64 + i];
    float Sf = f0 / (f0 + run);
    fin[((size_t)bn * CIN + 2 * CV + i) * HW + hw]        = (bf16_t)Sf;
    fin[((size_t)bn * CIN + 2 * CV + TOPL + i) * HW + hw] = (bf16_t)(1.f - Sf);
  }
}

// ---------------------------------------------------------------------------
// K11: broadcast qv into fin channels [CV, 2*CV)
// ---------------------------------------------------------------------------
__global__ void k_qvcopy(const float* __restrict__ qv, bf16_t* __restrict__ fin) {
  size_t i = (size_t)blockIdx.x * blockDim.x + threadIdx.x;  // BN*CV*HW
  int hw = (int)(i & 4095);
  int cv = (int)((i >> 12) & 511);
  int bn = (int)(i >> 21);
  int b = bn >> 2;
  fin[((size_t)bn * CIN + CV + cv) * HW + hw] = (bf16_t)qv[((size_t)b * CV + cv) * HW + hw];
}

// ---------------------------------------------------------------------------
// K12: mem_out GEMM -> fin channels [0, CV).  M=512 N=4096 K=512 per bn.
// ---------------------------------------------------------------------------
__global__ __launch_bounds__(256) void k_memout(const bf16_t* __restrict__ mv2,
                                                const bf16_t* __restrict__ pafft,
                                                bf16_t* __restrict__ fin) {
  int wave = threadIdx.x >> 5, lane = threadIdx.x & 31;
  int bn = blockIdx.y;
  int g = blockIdx.x * 8 + wave;          // 0..2047
  int mt = g >> 6, ns = g & 63;
  const bf16_t* A  = mv2 + ((size_t)bn * CV + (size_t)mt * 16) * (2 * DL);
  const bf16_t* Bt = pafft + ((size_t)bn * HW + (size_t)ns * 64) * (2 * DL);
  v8f acc[4] = {};
  for (int kk = 0; kk < 2 * DL; kk += 32) {
    v16bf af = load_a_rm(A + kk, 2 * DL, lane);
#pragma unroll
    for (int t = 0; t < 4; ++t)
      acc[t] = wmma_bf16(af, load_b_tn(Bt + (size_t)t * 16 * (2 * DL) + kk, 2 * DL, lane), acc[t]);
  }
  int col = lane & 15, mh = lane >> 4;
#pragma unroll
  for (int t = 0; t < 4; ++t)
#pragma unroll
    for (int v = 0; v < 8; ++v)
      fin[((size_t)bn * CIN + mt * 16 + mh * 8 + v) * HW + ns * 64 + t * 16 + col] =
          (bf16_t)acc[t][v];
}

// ---------------------------------------------------------------------------
// K13: f32 -> bf16 convert (conv weights)
// ---------------------------------------------------------------------------
__global__ void k_cvt(const float* __restrict__ src, bf16_t* __restrict__ dst, int n) {
  int i = blockIdx.x * blockDim.x + threadIdx.x;
  if (i < n) dst[i] = (bf16_t)src[i];
}

// ---------------------------------------------------------------------------
// K14: im2col for one bn: bT(HW, KTOT) bf16, K-contiguous per pixel.
// ---------------------------------------------------------------------------
__global__ void k_im2col(const bf16_t* __restrict__ fin_bn, bf16_t* __restrict__ bT) {
  int idx = blockIdx.x * 256 + threadIdx.x;     // HW*CIN
  if (idx >= HW * CIN) return;
  int ci = idx % CIN, n = idx / CIN;
  int y = n >> 6, x = n & 63;
  const bf16_t* F = fin_bn + (size_t)ci * HW;
  bf16_t* o = bT + (size_t)n * KTOT + ci * 9;
#pragma unroll
  for (int ky = 0; ky < 3; ++ky) {
    int yy = y + ky - 1;
#pragma unroll
    for (int kx = 0; kx < 3; ++kx) {
      int xx = x + kx - 1;
      o[ky * 3 + kx] = (yy >= 0 && yy < DH && xx >= 0 && xx < DW)
                           ? F[yy * DW + xx] : (bf16_t)0.f;
    }
  }
}

// ---------------------------------------------------------------------------
// K15: gated conv GEMM for one bn. Block tile 128(M) x 64(N):
// 8 waves share the ns strip; the 64x32 B chunk (4KB) is double-buffered in
// LDS via GLOBAL_LOAD_ASYNC_TO_LDS_B128 (each of 256 lanes copies 16B), so B
// L2 traffic drops 8x and the async stage of chunk k+1 overlaps the WMMAs of
// chunk k. A (Wf/Wa rows) streams from global with wide b128 loads.
// grid(64 ns, 4 mb), block 256.
// ---------------------------------------------------------------------------
__global__ __launch_bounds__(256) void k_convgemm(const bf16_t* __restrict__ wf,
                                                  const bf16_t* __restrict__ wa,
                                                  const bf16_t* __restrict__ bT,
                                                  const float* __restrict__ bfv,
                                                  const float* __restrict__ bav,
                                                  float* __restrict__ out_bn) {
  __shared__ bf16_t lds[2][64 * 32];            // 2 x 4KB double buffer
  int tid = threadIdx.x;
  int wave = tid >> 5, lane = tid & 31;
  int ns = blockIdx.x;                          // 0..63  (N strip of 64)
  int mt = blockIdx.y * 8 + wave;               // 0..31  (M tile of 16)
  const bf16_t* Af = wf + (size_t)mt * 16 * KTOT;
  const bf16_t* Aa = wa + (size_t)mt * 16 * KTOT;
  const bf16_t* Bt = bT + (size_t)ns * 64 * KTOT;
  int srow = tid >> 2, spart = tid & 3;         // 64 rows x 4 x 16B = 4KB
  const bf16_t* gsrc = Bt + (size_t)srow * KTOT + spart * 8;
  unsigned ldst0 = (unsigned)(size_t)&lds[0][srow * 32 + spart * 8];
  unsigned ldst1 = (unsigned)(size_t)&lds[1][srow * 32 + spart * 8];
  // stage chunk 0
  async_copy16(ldst0, gsrc);
  v8f accf[4] = {}, acca[4] = {};
  for (int kk = 0; kk < KTOT; kk += 32) {
    int cur = (kk >> 5) & 1;
    wait_async0();
    __syncthreads();                            // chunk `cur` visible to all
    if (kk + 32 < KTOT)                         // stage next chunk (overlaps)
      async_copy16(cur ? ldst0 : ldst1, gsrc + kk + 32);
    __builtin_prefetch(Af + kk + 256, 0, 1);
    __builtin_prefetch(Aa + kk + 256, 0, 1);
    v16bf af = load_a_rm(Af + kk, KTOT, lane);
    v16bf aa = load_a_rm(Aa + kk, KTOT, lane);
    const bf16_t* Lb = &lds[cur][0];
#pragma unroll
    for (int t = 0; t < 4; ++t) {
      v16bf bfr = load_b_lds(Lb + t * 16 * 32, lane);
      accf[t] = wmma_bf16(af, bfr, accf[t]);
      acca[t] = wmma_bf16(aa, bfr, acca[t]);
    }
  }
  int col = lane & 15, mh = lane >> 4;
#pragma unroll
  for (int t = 0; t < 4; ++t)
#pragma unroll
    for (int v = 0; v < 8; ++v) {
      int m = mt * 16 + mh * 8 + v;
      int n = ns * 64 + t * 16 + col;
      float fv = accf[t][v] + bfv[m];
      float av = acca[t][v] + bav[m];
      out_bn[(size_t)m * HW + n] = fv * (1.f / (1.f + __expf(-av)));
    }
}

// ---------------------------------------------------------------------------
// host launcher
// ---------------------------------------------------------------------------
extern "C" void kernel_launch(void* const* d_in, const int* in_sizes, int n_in,
                              void* d_out, int out_size, void* d_ws, size_t ws_size,
                              hipStream_t stream) {
  (void)in_sizes; (void)n_in; (void)out_size; (void)ws_size;
  const float* qk     = (const float*)d_in[0];
  const float* qv     = (const float*)d_in[1];
  const float* qv_mem = (const float*)d_in[2];
  const float* masks  = (const float*)d_in[3];
  const float* kappa0 = (const float*)d_in[4];
  // d_in[5] = nu0 (zeros, algebraically drops out)
  const float* zita0  = (const float*)d_in[6];
  const float* Wf     = (const float*)d_in[7];
  const float* bfv    = (const float*)d_in[8];
  const float* Wa     = (const float*)d_in[9];
  const float* bav    = (const float*)d_in[10];
  float* out = (float*)d_out;

  char* ws = (char*)d_ws;
  bf16_t* xt    = (bf16_t*)(ws + OFF_XT);
  bf16_t* xtn   = (bf16_t*)(ws + OFF_XTN);
  bf16_t* xb    = (bf16_t*)(ws + OFF_XB);
  bf16_t* knt   = (bf16_t*)(ws + OFF_KNT);
  float*  kappa = (float*) (ws + OFF_KAPPA);
  float*  wei   = (float*) (ws + OFF_WEI);
  float*  zita  = (float*) (ws + OFF_ZITA);
  float*  raw   = (float*) (ws + OFF_RAW);
  bf16_t* zzt   = (bf16_t*)(ws + OFF_ZZB);   // zz^T, later paff^T
  bf16_t* mv2   = (bf16_t*)(ws + OFF_MV2);
  bf16_t* fin   = (bf16_t*)(ws + OFF_FIN);
  bf16_t* wfb   = (bf16_t*)(ws + OFF_WFB);
  bf16_t* wab   = (bf16_t*)(ws + OFF_WAB);
  bf16_t* bT    = (bf16_t*)(ws + OFF_IM2);   // overlays raw/zzt (dead by conv)

  // prep
  k_prep<<<(DB * HW) / 256, 256, 0, stream>>>(qk, xt, xtn, xb);
  k_copy_f32<<<(BNS * CK * DL) / 256, 256, 0, stream>>>(kappa0, kappa, BNS * CK * DL);
  k_copy_f32<<<(BNS * HW) / 256, 256, 0, stream>>>(masks, wei, BNS * HW);

  // EM iterations
  for (int it = 0; it < 4; ++it) {
    k_knorm<<<(BNS * DL) / 256, 256, 0, stream>>>(kappa, knt);
    k_gemm1<<<dim3(128, BNS), 256, 0, stream>>>(xt, knt, raw);
    k_softmax_zz<<<(BNS * HW) / 256, 256, 0, stream>>>(raw, wei, zzt);
    k_zita<<<dim3(DL, BNS), 256, 0, stream>>>(zzt, zita0, zita);
    k_kappa<<<dim3(8, BNS), 256, 0, stream>>>(xb, zzt, kappa0, zita0, zita, kappa);
    if (it < 3) {
      k_knorm<<<(BNS * DL) / 256, 256, 0, stream>>>(kappa, knt);
      k_gemm1<<<dim3(128, BNS), 256, 0, stream>>>(xtn, knt, raw);
      k_sww<<<(BN * HW) / 256, 256, 0, stream>>>(raw, masks, wei);
    }
  }

  // nu -> mv2 (consumes zz^T before buffer is reused for paff^T)
  k_nu<<<dim3(64, BNS), 256, 0, stream>>>(qv_mem, zzt, zita, mv2);

  // matching / affinity
  k_knorm<<<(BNS * DL) / 256, 256, 0, stream>>>(kappa, knt);
  k_aff<<<dim3(128, BNS), 256, 0, stream>>>(knt, xtn, raw);
  k_paff<<<dim3(HW / 256, BN), 256, 0, stream>>>(raw, zzt);
  k_topk<<<(BN * HW) / 16, 16, 0, stream>>>(raw, fin);

  // assemble fin and mem_out
  k_qvcopy<<<(int)(((size_t)BN * CV * HW) / 256), 256, 0, stream>>>(qv, fin);
  k_memout<<<dim3(256, BN), 256, 0, stream>>>(mv2, zzt, fin);

  // gated conv fusion: per-bn im2col + double GEMM (im2col buffer L2-resident)
  k_cvt<<<(CV * KTOT) / 256, 256, 0, stream>>>(Wf, wfb, CV * KTOT);
  k_cvt<<<(CV * KTOT) / 256, 256, 0, stream>>>(Wa, wab, CV * KTOT);
  for (int bn = 0; bn < BN; ++bn) {
    k_im2col<<<(HW * CIN + 255) / 256, 256, 0, stream>>>(fin + (size_t)bn * CIN * HW, bT);
    k_convgemm<<<dim3(64, 4), 256, 0, stream>>>(wfb, wab, bT, bfv, bav,
                                                out + (size_t)bn * CV * HW);
  }
}
